// TGATCell_7215545057459
// MI455X (gfx1250) — matrix operand order repaired and checked by reading, not compiled, not run on hardware
//
#include <hip/hip_runtime.h>
#include <hip/hip_bf16.h>
#include <math.h>
#include <stdint.h>

typedef __attribute__((ext_vector_type(16))) _Float16 v16h;
typedef __attribute__((ext_vector_type(8)))  float    v8f;
typedef unsigned int u32x4 __attribute__((ext_vector_type(4)));
typedef int          i32x4 __attribute__((ext_vector_type(4)));
typedef int          i32x8 __attribute__((ext_vector_type(8)));

#define IN_DIM    128
#define OUT_DIM   128
#define NHEADS    4
#define NEG_SLOPE 0.2f
#define ASTRIDE   136   // LDS row stride in halves: 272B = 16B-aligned, banks 4*r distinct

#if __has_builtin(__builtin_amdgcn_tensor_load_to_lds) && __has_builtin(__builtin_amdgcn_s_wait_tensorcnt)
#define HAVE_TDM 1
#else
#define HAVE_TDM 0
#endif

// ---------- helpers ----------

__device__ __forceinline__ float sigf(float x) { return 1.0f / (1.0f + __expf(-x)); }

// monotone float -> uint encoding so unsigned atomicMax == float max
__device__ __forceinline__ unsigned encf(float f) {
    unsigned u = __float_as_uint(f);
    return (u & 0x80000000u) ? ~u : (u | 0x80000000u);
}
__device__ __forceinline__ float decf(unsigned u) {
    return (u & 0x80000000u) ? __uint_as_float(u ^ 0x80000000u) : __uint_as_float(~u);
}

// A fragment (16x32 f16, ISA 7.12.2 layout) from LDS, row stride ASTRIDE halves.
__device__ __forceinline__ v16h load_a_frag(const _Float16* lds, int k0) {
    const int lane = threadIdx.x & 31;
    const int r  = lane & 15;     // M row
    const int hs = lane >> 4;     // K half select
    v16h a{};
#pragma unroll
    for (int v = 0; v < 8; ++v) {
        const int kb = k0 + ((v < 4) ? (2 * v) : (16 + 2 * (v - 4))) + 8 * hs;
        a[2 * v]     = lds[r * ASTRIDE + kb];
        a[2 * v + 1] = lds[r * ASTRIDE + kb + 1];
    }
    return a;
}

// TDM: async 2-D f16 tile (rows x cols) Global -> LDS with LDS row padding to 272B.
// D# layout per CDNA5 ISA.
__device__ __forceinline__ void tdm_tile_f16(const _Float16* lds_dst,
                                             const _Float16* gsrc,
                                             int ld_elems, int rows, int cols) {
#if HAVE_TDM
    unsigned lds_byte = (unsigned)(size_t)lds_dst;             // low 32b of flat = LDS byte addr
    unsigned long long ga = (unsigned long long)(size_t)gsrc;  // tile start
    u32x4 g0;
    g0.x = 1u;                                                 // count=1, user descriptor
    g0.y = lds_byte;                                           // lds_addr
    g0.z = (unsigned)ga;                                       // global_addr[31:0]
    g0.w = ((unsigned)(ga >> 32) & 0x01FFFFFFu) | 0x80000000u; // global_addr[56:32] | type=2
    i32x8 g1;
    // data_size=1 (2B) | pad_enable | pad_interval=5 (64 dwords=256B) | pad_amount=3 (4 dwords=16B)
    g1[0] = (int)((1u << 16) | (1u << 20) | (5u << 22) | (3u << 25));
    g1[1] = (int)(((unsigned)cols & 0xFFFFu) << 16);                              // tensor_dim0 lo16
    g1[2] = (int)((((unsigned)cols) >> 16) | (((unsigned)rows & 0xFFFFu) << 16)); // dim0 hi | dim1 lo
    g1[3] = (int)((((unsigned)rows) >> 16) | (((unsigned)cols) << 16));           // dim1 hi | tile_dim0
    g1[4] = rows;                                                                 // tile_dim1 (tile_dim2=0)
    g1[5] = ld_elems;                                                             // tensor_dim0_stride lo
    g1[6] = 0;
    g1[7] = 0;
    i32x4 g2 = {0, 0, 0, 0};
    i32x4 g3 = {0, 0, 0, 0};
#if defined(__clang_major__) && __clang_major__ >= 23
    i32x8 gz = {0, 0, 0, 0, 0, 0, 0, 0};
    __builtin_amdgcn_tensor_load_to_lds(g0, g1, g2, g3, gz, 0);
#else
    __builtin_amdgcn_tensor_load_to_lds(g0, g1, g2, g3, 0);
#endif
#else
    (void)lds_dst; (void)gsrc; (void)ld_elems; (void)rows; (void)cols;
#endif
}

// ---------- kernels ----------

__global__ void __launch_bounds__(256) zero_f32(float* __restrict__ p, int count) {
    int t = blockIdx.x * 256 + threadIdx.x;
    if (t < count) p[t] = 0.0f;
}

__global__ void __launch_bounds__(256)
cvt_f16(const float* __restrict__ src, _Float16* __restrict__ dst, int count) {
    int t = blockIdx.x * 256 + threadIdx.x;
    if (t < count) dst[t] = (_Float16)src[t];
}

// One-time weight swizzle: row-major f32 W[ K x ldb ] -> per-fragment f16 B layout.
// Fragment (kc,ct): 32 lanes x 16 halves; lane = kh*16+col holds (W[k][c], W[k+1][c]) pairs.
__global__ void __launch_bounds__(256)
swizzle_w(const float* __restrict__ W, int ldb, int kchunks, int ctiles,
          _Float16* __restrict__ dst) {
    int t = blockIdx.x * 256 + threadIdx.x;
    int total = kchunks * ctiles * 32;
    if (t >= total) return;
    int lane = t & 31;
    int frag = t >> 5;
    int ct = frag % ctiles, kc = frag / ctiles;
    int col = lane & 15, kh = lane >> 4;
    v16h b{};
#pragma unroll
    for (int v = 0; v < 8; ++v) {
        int k = kc * 32 + ((v < 4) ? (2 * v) : (16 + 2 * (v - 4))) + 8 * kh;
        b[2 * v]     = (_Float16)W[(size_t)k * ldb + ct * 16 + col];
        b[2 * v + 1] = (_Float16)W[(size_t)(k + 1) * ldb + ct * 16 + col];
    }
    *(v16h*)(dst + (size_t)t * 16) = b;
}

// out[N,128] = X16[N,128] @ W (pre-swizzled fragments), f16 WMMA, f32 acc
__global__ void __launch_bounds__(256)
xform_wmma(const _Float16* __restrict__ X16, const v16h* __restrict__ Bf,
           float* __restrict__ out, int n) {
    __shared__ _Float16 As[16 * ASTRIDE];
    const int row0 = blockIdx.x * 16;
    const int tid  = threadIdx.x;
#if HAVE_TDM
    if (tid < 32) {
        tdm_tile_f16(As, X16 + (size_t)row0 * IN_DIM, IN_DIM, 16, IN_DIM);
        __builtin_amdgcn_s_wait_tensorcnt(0);
    }
#else
#pragma unroll
    for (int i = 0; i < 8; ++i) {
        int idx = tid + i * 256;
        int r = idx >> 7, c = idx & 127;
        int gr = row0 + r; if (gr >= n) gr = n - 1;
        As[r * ASTRIDE + c] = X16[(size_t)gr * IN_DIM + c];
    }
#endif
    __syncthreads();
    const int wave = tid >> 5, lane = tid & 31;
    v8f acc = {};
#pragma unroll
    for (int kc = 0; kc < 4; ++kc) {
        if (kc < 3) __builtin_prefetch((const void*)&Bf[((kc + 1) * 8 + wave) * 32 + lane], 0, 1);
        v16h a = load_a_frag(As, kc * 32);
        v16h b = Bf[(kc * 8 + wave) * 32 + lane];
        acc = __builtin_amdgcn_wmma_f32_16x16x32_f16(false, a, false, b,
                                                     (short)0, acc, false, false);
    }
    const int nI = lane & 15, hs = lane >> 4;
    const int c0 = wave * 16;
    float* op = out + (size_t)(row0 + 8 * hs) * OUT_DIM + c0 + nI;
    if (row0 + 16 <= n) {          // uniform fast path: full tile, no per-element guards
#pragma unroll
        for (int v = 0; v < 8; ++v) op[(size_t)v * OUT_DIM] = acc[v];
    } else {
#pragma unroll
        for (int v = 0; v < 8; ++v) {
            if (row0 + 8 * hs + v < n) op[(size_t)v * OUT_DIM] = acc[v];
        }
    }
}

// one wave per edge: e[edge][head] = leaky(xl[src]+xr[dst]) . att ; atomicMax per dst
__global__ void __launch_bounds__(256)
edge_scores(const int* __restrict__ ei, int E, int Et,
            const float* __restrict__ xl, const float* __restrict__ xr,
            const float* __restrict__ att, float* __restrict__ ebuf,
            unsigned* __restrict__ nmax) {
    int wid = (blockIdx.x * 256 + threadIdx.x) >> 5;
    if (wid >= Et) return;
    const int lane = threadIdx.x & 31;
    int src, dst;
    if (wid < E) { src = ei[wid]; dst = ei[E + wid]; }
    else         { src = wid - E; dst = src; }
    float s = 0.0f;
#pragma unroll
    for (int j = 0; j < 4; ++j) {
        int idx = lane * 4 + j;                   // == head*32 + d
        float m  = xl[(size_t)src * OUT_DIM + idx] + xr[(size_t)dst * OUT_DIM + idx];
        float lm = m > 0.0f ? m : NEG_SLOPE * m;
        s += lm * att[idx];
    }
    s += __shfl_xor(s, 1);
    s += __shfl_xor(s, 2);
    s += __shfl_xor(s, 4);                        // sum within 8-lane head group
    if ((lane & 7) == 0) {
        int head = lane >> 3;
        ebuf[(size_t)wid * NHEADS + head] = s;
        atomicMax(&nmax[dst * NHEADS + head], encf(s));
    }
}

// a = exp(e - max[dst]); atomicAdd denom
__global__ void __launch_bounds__(256)
edge_exp(const int* __restrict__ ei, int E, int Et,
         float* __restrict__ ebuf, const unsigned* __restrict__ nmax,
         float* __restrict__ denom) {
    int t = blockIdx.x * 256 + threadIdx.x;
    if (t >= Et * NHEADS) return;
    int eid = t >> 2, head = t & 3;
    int dst = (eid < E) ? ei[E + eid] : (eid - E);
    float mx = decf(nmax[dst * NHEADS + head]);
    float a  = __expf(ebuf[t] - mx);
    ebuf[t]  = a;
    atomicAdd(&denom[dst * NHEADS + head], a);
}

// one wave per edge: agg[dst] += xl[src] * alpha
__global__ void __launch_bounds__(256)
edge_aggregate(const int* __restrict__ ei, int E, int Et,
               const float* __restrict__ xl, const float* __restrict__ ebuf,
               const float* __restrict__ denom, float* __restrict__ agg) {
    int wid = (blockIdx.x * 256 + threadIdx.x) >> 5;
    if (wid >= Et) return;
    const int lane = threadIdx.x & 31;
    int src, dst;
    if (wid < E) { src = ei[wid]; dst = ei[E + wid]; }
    else         { src = wid - E; dst = src; }
    int head = lane >> 3;
    float alpha = ebuf[(size_t)wid * NHEADS + head] /
                  (denom[dst * NHEADS + head] + 1e-16f);
#pragma unroll
    for (int j = 0; j < 4; ++j) {
        int idx = lane * 4 + j;
        atomicAdd(&agg[(size_t)dst * OUT_DIM + idx],
                  xl[(size_t)src * OUT_DIM + idx] * alpha);
    }
}

// f16 f = sigmoid(agg + bias)
__global__ void __launch_bounds__(256)
node_sigmoid(const float* __restrict__ agg, const float* __restrict__ bias,
             _Float16* __restrict__ f16out, int n) {
    int t = blockIdx.x * 256 + threadIdx.x;
    if (t >= n * OUT_DIM) return;
    f16out[t] = (_Float16)sigf(agg[t] + bias[t & (OUT_DIM - 1)]);
}

// ru[N,256] = sigmoid([f,h] @ W1 + b1)   (grid.y selects column half)
__global__ void __launch_bounds__(256)
gemm_ru(const _Float16* __restrict__ F16, const _Float16* __restrict__ H16,
        const v16h* __restrict__ Bf, const float* __restrict__ b1,
        float* __restrict__ ru, int n) {
    __shared__ _Float16 Af[16 * ASTRIDE];
    __shared__ _Float16 Ah[16 * ASTRIDE];
    const int row0 = blockIdx.x * 16;
    const int tid  = threadIdx.x;
#if HAVE_TDM
    if (tid < 32) {
        tdm_tile_f16(Af, F16 + (size_t)row0 * 128, 128, 16, 128);
        tdm_tile_f16(Ah, H16 + (size_t)row0 * 128, 128, 16, 128);
        __builtin_amdgcn_s_wait_tensorcnt(0);
    }
#else
#pragma unroll
    for (int i = 0; i < 8; ++i) {
        int idx = tid + i * 256;
        int r = idx >> 7, c = idx & 127;
        int gr = row0 + r; if (gr >= n) gr = n - 1;
        Af[r * ASTRIDE + c] = F16[(size_t)gr * 128 + c];
        Ah[r * ASTRIDE + c] = H16[(size_t)gr * 128 + c];
    }
#endif
    __syncthreads();
    const int wave = tid >> 5, lane = tid & 31;
    const int ct = blockIdx.y * 8 + wave;         // 16 column tiles total
    v8f acc = {};
#pragma unroll
    for (int kc = 0; kc < 8; ++kc) {
        if (kc < 7) __builtin_prefetch((const void*)&Bf[((kc + 1) * 16 + ct) * 32 + lane], 0, 1);
        v16h a = (kc < 4) ? load_a_frag(Af, kc * 32) : load_a_frag(Ah, (kc - 4) * 32);
        v16h b = Bf[(kc * 16 + ct) * 32 + lane];
        acc = __builtin_amdgcn_wmma_f32_16x16x32_f16(false, a, false, b,
                                                     (short)0, acc, false, false);
    }
    const int nI = lane & 15, hs = lane >> 4;
    const int c0 = ct * 16;
    const float bias = b1[c0 + nI];
    float* op = ru + (size_t)(row0 + 8 * hs) * 256 + c0 + nI;
    if (row0 + 16 <= n) {          // uniform fast path
#pragma unroll
        for (int v = 0; v < 8; ++v) op[(size_t)v * 256] = sigf(acc[v] + bias);
    } else {
#pragma unroll
        for (int v = 0; v < 8; ++v) {
            if (row0 + 8 * hs + v < n) op[(size_t)v * 256] = sigf(acc[v] + bias);
        }
    }
}

// c = tanh([f, r*h] @ W2 + b2); out = u*h + (1-u)*c
__global__ void __launch_bounds__(256)
gemm_c_final(const _Float16* __restrict__ F16, const float* __restrict__ h,
             const float* __restrict__ ru, const v16h* __restrict__ Bf,
             const float* __restrict__ b2, float* __restrict__ out, int n) {
    __shared__ _Float16 Af[16 * ASTRIDE];
    __shared__ _Float16 Ah[16 * ASTRIDE];
    const int row0 = blockIdx.x * 16;
    const int tid  = threadIdx.x;
#if HAVE_TDM
    if (tid < 32) tdm_tile_f16(Af, F16 + (size_t)row0 * 128, 128, 16, 128);
#endif
    // r*h half is computed on the fly (needs f32 ru*h product), staged manually
#pragma unroll
    for (int i = 0; i < 8; ++i) {
        int idx = tid + i * 256;
        int r = idx >> 7, c = idx & 127;
        int gr = row0 + r; if (gr >= n) gr = n - 1;
        Ah[r * ASTRIDE + c] = (_Float16)(ru[(size_t)gr * 256 + c] * h[(size_t)gr * 128 + c]);
#if !HAVE_TDM
        Af[r * ASTRIDE + c] = F16[(size_t)gr * 128 + c];
#endif
    }
#if HAVE_TDM
    if (tid < 32) __builtin_amdgcn_s_wait_tensorcnt(0);
#endif
    __syncthreads();
    const int wave = tid >> 5, lane = tid & 31;
    v8f acc = {};
#pragma unroll
    for (int kc = 0; kc < 8; ++kc) {
        if (kc < 7) __builtin_prefetch((const void*)&Bf[((kc + 1) * 8 + wave) * 32 + lane], 0, 1);
        v16h a = (kc < 4) ? load_a_frag(Af, kc * 32) : load_a_frag(Ah, (kc - 4) * 32);
        v16h b = Bf[(kc * 8 + wave) * 32 + lane];
        acc = __builtin_amdgcn_wmma_f32_16x16x32_f16(false, a, false, b,
                                                     (short)0, acc, false, false);
    }
    const int nI = lane & 15, hs = lane >> 4;
    const int c0 = wave * 16;
    const int col = c0 + nI;
    const int rbase = row0 + 8 * hs;
    const float bias = b2[col];
    if (row0 + 16 <= n) {          // uniform fast path
#pragma unroll
        for (int v = 0; v < 8; ++v) {
            size_t gr = (size_t)(rbase + v);
            float u  = ru[gr * 256 + 128 + col];
            float hv = h[gr * 128 + col];
            float cv = tanhf(acc[v] + bias);
            out[gr * 128 + col] = u * hv + (1.0f - u) * cv;
        }
    } else {
#pragma unroll
        for (int v = 0; v < 8; ++v) {
            if (rbase + v < n) {
                size_t gr = (size_t)(rbase + v);
                float u  = ru[gr * 256 + 128 + col];
                float hv = h[gr * 128 + col];
                float cv = tanhf(acc[v] + bias);
                out[gr * 128 + col] = u * hv + (1.0f - u) * cv;
            }
        }
    }
}

// ---------- launcher ----------

extern "C" void kernel_launch(void* const* d_in, const int* in_sizes, int n_in,
                              void* d_out, int out_size, void* d_ws, size_t ws_size,
                              hipStream_t stream) {
    const float* x    = (const float*)d_in[0];
    const int*   ei   = (const int*)d_in[1];
    // d_in[2] = edge_weight (unused by reference)
    const float* h    = (const float*)d_in[3];
    const float* Wl   = (const float*)d_in[4];
    const float* Wr   = (const float*)d_in[5];
    const float* att  = (const float*)d_in[6];
    const float* bc   = (const float*)d_in[7];
    const float* W1   = (const float*)d_in[8];
    const float* b1   = (const float*)d_in[9];
    const float* W2   = (const float*)d_in[10];
    const float* b2   = (const float*)d_in[11];
    float* out = (float*)d_out;

    const int N  = in_sizes[0] / IN_DIM;   // 50000
    const int E  = in_sizes[1] / 2;        // 800000
    const int Et = E + N;                  // with self-loops

    // workspace layout (float units, 16-float (64B) aligned chunks)
    float* ws = (float*)d_ws;
    size_t off = 0;
#define TAKE(nfloats) (ws + off); off = (off + (size_t)(nfloats) + 15) & ~(size_t)15
    float*    XL   = TAKE((size_t)N * OUT_DIM);
    float*    XR   = TAKE((size_t)N * OUT_DIM);
    float*    EB   = TAKE((size_t)Et * NHEADS);
    float*    AGG  = TAKE((size_t)N * OUT_DIM + 2 * (size_t)N * NHEADS); // AGG|DEN|NMAX contiguous
    float*    DEN  = AGG + (size_t)N * OUT_DIM;
    unsigned* NMAX = (unsigned*)(DEN + (size_t)N * NHEADS);
    float*    RU   = TAKE((size_t)N * 256);
    _Float16* X16  = (_Float16*)TAKE((size_t)N * IN_DIM / 2);
    _Float16* F16  = (_Float16*)TAKE((size_t)N * OUT_DIM / 2);
    _Float16* H16  = (_Float16*)TAKE((size_t)N * OUT_DIM / 2);
    _Float16* WlF  = (_Float16*)TAKE(4 * 8 * 32 * 16 / 2);
    _Float16* WrF  = (_Float16*)TAKE(4 * 8 * 32 * 16 / 2);
    _Float16* W1F  = (_Float16*)TAKE(8 * 16 * 32 * 16 / 2);
    _Float16* W2F  = (_Float16*)TAKE(8 * 8 * 32 * 16 / 2);
#undef TAKE
    (void)ws_size; (void)n_in; (void)out_size;

    const int rowTiles = (N + 15) / 16;

    // precision/layout prep: f16 copies of x, h; weights into WMMA B-fragment layout
    cvt_f16<<<(N * IN_DIM + 255) / 256, 256, 0, stream>>>(x, X16, N * IN_DIM);
    cvt_f16<<<(N * OUT_DIM + 255) / 256, 256, 0, stream>>>(h, H16, N * OUT_DIM);
    swizzle_w<<<(4 * 8 * 32 + 255) / 256, 256, 0, stream>>>(Wl, 128, 4, 8, WlF);
    swizzle_w<<<(4 * 8 * 32 + 255) / 256, 256, 0, stream>>>(Wr, 128, 4, 8, WrF);
    swizzle_w<<<(8 * 16 * 32 + 255) / 256, 256, 0, stream>>>(W1, 256, 8, 16, W1F);
    swizzle_w<<<(8 * 8 * 32 + 255) / 256, 256, 0, stream>>>(W2, 128, 8, 8, W2F);

    // zero AGG + DEN + NMAX in one shot (contiguous; 0u == float 0 == enc floor)
    int zc = N * (OUT_DIM + 2 * NHEADS);
    zero_f32<<<(zc + 255) / 256, 256, 0, stream>>>(AGG, zc);

    // node transforms (TDM-staged A tiles, WMMA)
    xform_wmma<<<rowTiles, 256, 0, stream>>>(X16, (const v16h*)WlF, XL, N);
    xform_wmma<<<rowTiles, 256, 0, stream>>>(X16, (const v16h*)WrF, XR, N);

    // edge softmax-attention passes (L2-resident gathers + atomics)
    int eBlk = (Et + 7) / 8;  // 8 waves per block
    edge_scores<<<eBlk, 256, 0, stream>>>(ei, E, Et, XL, XR, att, EB, NMAX);
    edge_exp<<<(Et * NHEADS + 255) / 256, 256, 0, stream>>>(ei, E, Et, EB, NMAX, DEN);
    edge_aggregate<<<eBlk, 256, 0, stream>>>(ei, E, Et, XL, EB, DEN, AGG);

    // f = sigmoid(agg + bias) -> f16
    node_sigmoid<<<(N * OUT_DIM + 255) / 256, 256, 0, stream>>>(AGG, bc, F16, N);

    // GRU cell (WMMA)
    gemm_ru<<<dim3(rowTiles, 2), 256, 0, stream>>>(F16, H16, (const v16h*)W1F, b1, RU, N);
    gemm_c_final<<<rowTiles, 256, 0, stream>>>(F16, h, RU, (const v16h*)W2F, b2, out, N);
}
